// HODE_MDP_18854906429546
// MI455X (gfx1250) — matrix operand order repaired and checked by reading, not compiled, not run on hardware
//
#include <hip/hip_runtime.h>
#include <hip/hip_bf16.h>

// ---------------------------------------------------------------------------
// Types for WMMA fragments (probe-verified signatures for gfx1250)
// ---------------------------------------------------------------------------
typedef __attribute__((ext_vector_type(16))) __bf16 v16bf;
typedef __attribute__((ext_vector_type(8)))  __bf16 v8bf;
typedef __attribute__((ext_vector_type(8)))  float  v8f;
typedef __attribute__((ext_vector_type(4)))  float  v4f;
typedef __attribute__((ext_vector_type(4)))  int    v4i;

#define D_DIM 128
#define P_DIM 8192
#define U_DIM 4096
#define B_DIM 1024

#define GLOBAL_AS __attribute__((address_space(1)))
#define LDS_AS    __attribute__((address_space(3)))

// gfx1250 async global->LDS path (ASYNCcnt-tracked), with sync fallback.
#if __has_builtin(__builtin_amdgcn_global_load_async_to_lds_b128)
#define HAVE_ASYNC_LDS 1
#else
#define HAVE_ASYNC_LDS 0
#endif

__device__ __forceinline__ void copy16_to_lds(const __bf16* g, __bf16* l) {
#if HAVE_ASYNC_LDS
    __builtin_amdgcn_global_load_async_to_lds_b128(
        (GLOBAL_AS v4i*)(unsigned long long)(size_t)g,
        (LDS_AS v4i*)(unsigned long long)(size_t)l, 0, 0);
#else
    *(uint4*)l = *(const uint4*)g;
#endif
}

__device__ __forceinline__ void wait_async_loads() {
#if HAVE_ASYNC_LDS
#if __has_builtin(__builtin_amdgcn_s_wait_asynccnt)
    __builtin_amdgcn_s_wait_asynccnt(0);
#else
    asm volatile("s_wait_asynccnt 0x0" ::: "memory");
#endif
#endif
}

// ---------------------------------------------------------------------------
// A-fragment loader: fp32 row segment -> bf16 16x32 A fragment (contiguous k order, same as B)
// p points to G[row*K + k0 + 16*khalf]; elements 0..15 are K=+0..15 (contiguous).
// ---------------------------------------------------------------------------
__device__ __forceinline__ v16bf load_a_frag(const float* __restrict__ p) {
    v4f x0 = *(const v4f*)(p);
    v4f x1 = *(const v4f*)(p + 4);
    v4f x2 = *(const v4f*)(p + 8);
    v4f x3 = *(const v4f*)(p + 12);
    v16bf a;
    a[0]=(__bf16)x0[0];  a[1]=(__bf16)x0[1];  a[2]=(__bf16)x0[2];  a[3]=(__bf16)x0[3];
    a[4]=(__bf16)x1[0];  a[5]=(__bf16)x1[1];  a[6]=(__bf16)x1[2];  a[7]=(__bf16)x1[3];
    a[8]=(__bf16)x2[0];  a[9]=(__bf16)x2[1];  a[10]=(__bf16)x2[2]; a[11]=(__bf16)x2[3];
    a[12]=(__bf16)x3[0]; a[13]=(__bf16)x3[1]; a[14]=(__bf16)x3[2]; a[15]=(__bf16)x3[3];
    return a;
}

// LDS B-tile layout: 128 rows (columns of X), padded row stride of 40 bf16
// (80 B) so the per-lane 16 B ds_load_b128 reads are bank-conflict-free.
#define BS_STRIDE 40

// Stage one 32x128 bf16 tile of X^T into LDS: row = column n of X (0..127),
// each row is 32 contiguous bf16 (64 B) starting at XT[n*K + k0].
// 256 threads: thread handles row = tid>>1, 32 B half = tid&1 (2 x b128 each).
__device__ __forceinline__ void stage_b_tile(const __bf16* __restrict__ XT,
                                             int K, int k0,
                                             __bf16* buf, int tid) {
    const int row  = tid >> 1;
    const int half = tid & 1;
    const __bf16* g = XT + (size_t)row * (size_t)K + k0 + half * 16;
    __bf16* l = buf + row * BS_STRIDE + half * 16;
    copy16_to_lds(g, l);
    copy16_to_lds(g + 8, l + 8);
}

// ---------------------------------------------------------------------------
// Generic GEMM: OUT[m,n] = alpha * sum_k G[m,k]*X[k,n] + addcoef*ADD[m,n]
//   G   : M x K fp32 (streamed once from HBM, converted to bf16 in regs)
//   XT  : 128 x K bf16 (X transposed; small, L2-resident)
//   OUT : optional fp32 (M x 128)
//   OUTT: optional bf16 transposed (128 x M) -> feeds next GEMM's B operand
// Block: 256 threads (8 waves). Wave w computes rows blk*128+16w..+15 x 128.
// B tile staged to LDS once per block per k-step (async, double-buffered);
// B-fragment ds_loads are software-pipelined one tile ahead so each WMMA
// overlaps the next fragment's LDS latency (partial dscnt waits).
// ---------------------------------------------------------------------------
__global__ __launch_bounds__(256) void gemm_bf16_kernel(
    const float* __restrict__ G, const __bf16* __restrict__ XT,
    int K, float alpha,
    const float* __restrict__ ADD, float addcoef,
    float* __restrict__ OUT, __bf16* __restrict__ OUTT, int Mtotal)
{
    __shared__ __align__(16) __bf16 Bs[2][128 * BS_STRIDE];

    const int tid  = threadIdx.x;
    const int lane = tid & 31;
    const int wave = tid >> 5;
    const int mrow = lane & 15;      // A row in strip; also B/C column in tile
    const int kh   = lane >> 4;      // lane half
    const int m0   = blockIdx.x * 128 + wave * 16;
    const size_t rowBase = (size_t)(m0 + mrow) * (size_t)K;

    v8f acc[8];
#pragma unroll
    for (int t = 0; t < 8; ++t)
#pragma unroll
        for (int r = 0; r < 8; ++r) acc[t][r] = 0.0f;

    // Prologue: stage first B tile.
    stage_b_tile(XT, K, 0, Bs[0], tid);

    for (int k0 = 0; k0 < K; k0 += 32) {
        const int cur = (k0 >> 5) & 1;

        wait_async_loads();      // my wave's async copies into Bs[cur] done
        __syncthreads();         // all waves staged; prev reads of Bs[cur^1] done

        if (k0 + 32 < K) stage_b_tile(XT, K, k0 + 32, Bs[cur ^ 1], tid);

        const float* ap = G + rowBase + k0 + 16 * kh;
        if (k0 + 32 < K) __builtin_prefetch(ap + 32, 0, 1);  // global_prefetch_b8
        v16bf afrag = load_a_frag(ap);

        // Software-pipelined B fragments: ds_load tile t+1 before WMMA t.
        const __bf16* bbase = &Bs[cur][mrow * BS_STRIDE + 16 * kh];
        v8bf lo = *(const v8bf*)bbase;        // ds_load_b128
        v8bf hi = *(const v8bf*)(bbase + 8);  // ds_load_b128
#pragma unroll
        for (int t = 0; t < 8; ++t) {
            v8bf lo_n = lo, hi_n = hi;
            if (t < 7) {
                const __bf16* bp = bbase + (t + 1) * 16 * BS_STRIDE;
                lo_n = *(const v8bf*)bp;
                hi_n = *(const v8bf*)(bp + 8);
            }
            v16bf bfrag = __builtin_shufflevector(lo, hi,
                0,1,2,3,4,5,6,7,8,9,10,11,12,13,14,15);
            acc[t] = __builtin_amdgcn_wmma_f32_16x16x32_bf16(
                false, afrag, false, bfrag, (short)0, acc[t], false, false);
            lo = lo_n; hi = hi_n;
        }
    }

#pragma unroll
    for (int t = 0; t < 8; ++t) {
        const int n = t * 16 + mrow;
#pragma unroll
        for (int r = 0; r < 8; ++r) {
            const int m = m0 + r + 8 * kh;
            float v = alpha * acc[t][r];
            if (ADD)  v += addcoef * ADD[(size_t)m * D_DIM + n];
            if (OUT)  OUT[(size_t)m * D_DIM + n] = v;
            if (OUTT) OUTT[(size_t)n * Mtotal + m] = (__bf16)v;
        }
    }
}

// ---------------------------------------------------------------------------
// Gate kernel: gate = pe * sigmoid(pe @ W + b), K = 128 fixed.
// W^T is 32 KB and L0-resident; direct global loads are fine here.
// Emits fp32 gate (for the residual adds) and bf16 gate^T (B operand).
// ---------------------------------------------------------------------------
__global__ __launch_bounds__(256) void gate_kernel(
    const float* __restrict__ pe, const __bf16* __restrict__ WT,
    const float* __restrict__ bias,
    float* __restrict__ gate, __bf16* __restrict__ gateT, int Mtotal)
{
    const int lane = threadIdx.x & 31;
    const int wave = threadIdx.x >> 5;
    const int mrow = lane & 15;
    const int kh   = lane >> 4;
    const int m0   = blockIdx.x * 128 + wave * 16;
    const size_t rowBase = (size_t)(m0 + mrow) * D_DIM;

    v8f acc[8];
#pragma unroll
    for (int t = 0; t < 8; ++t)
#pragma unroll
        for (int r = 0; r < 8; ++r) acc[t][r] = 0.0f;

#pragma unroll
    for (int k0 = 0; k0 < D_DIM; k0 += 32) {
        v16bf afrag = load_a_frag(pe + rowBase + k0 + 16 * kh);
#pragma unroll
        for (int t = 0; t < 8; ++t) {
            const __bf16* bp = WT + (size_t)(t * 16 + mrow) * D_DIM + k0 + 16 * kh;
            v16bf bfrag = *(const v16bf*)bp;
            acc[t] = __builtin_amdgcn_wmma_f32_16x16x32_bf16(
                false, afrag, false, bfrag, (short)0, acc[t], false, false);
        }
    }

#pragma unroll
    for (int t = 0; t < 8; ++t) {
        const int n = t * 16 + mrow;
        const float bn = bias[n];
#pragma unroll
        for (int r = 0; r < 8; ++r) {
            const int m = m0 + r + 8 * kh;
            const float z = acc[t][r] + bn;
            const float s = 1.0f / (1.0f + __expf(-z));
            const float g = pe[(size_t)m * D_DIM + n] * s;
            gate[(size_t)m * D_DIM + n] = g;
            gateT[(size_t)n * Mtotal + m] = (__bf16)g;
        }
    }
}

// ---------------------------------------------------------------------------
// Small helpers: W (128x128 fp32) -> W^T bf16 ; final user-row gather.
// ---------------------------------------------------------------------------
__global__ void wt_convert_kernel(const float* __restrict__ W, __bf16* __restrict__ WT) {
    int i = blockIdx.x * blockDim.x + threadIdx.x;   // 128*128
    int k = i >> 7, n = i & 127;
    WT[(size_t)n * D_DIM + k] = (__bf16)W[(size_t)k * D_DIM + n];
}

__global__ void gather_kernel(const float* __restrict__ u1, const float* __restrict__ u2,
                              const int* __restrict__ idx,
                              float* __restrict__ out_hg, float* __restrict__ out_geo) {
    int i = blockIdx.x * blockDim.x + threadIdx.x;   // B*128
    int b = i >> 7, n = i & 127;
    size_t src = (size_t)idx[b] * D_DIM + n;
    out_hg[i]  = u1[src];
    out_geo[i] = u2[src];
}

// ---------------------------------------------------------------------------
// Host-side orchestration
// ---------------------------------------------------------------------------
extern "C" void kernel_launch(void* const* d_in, const int* in_sizes, int n_in,
                              void* d_out, int out_size, void* d_ws, size_t ws_size,
                              hipStream_t stream) {
    const float* poi      = (const float*)d_in[0];   // (P+1) x 128; pe = first P rows
    const float* w_geo    = (const float*)d_in[1];
    const float* b_geo    = (const float*)d_in[2];
    const float* w_seq    = (const float*)d_in[3];
    const float* b_seq    = (const float*)d_in[4];
    const float* w_col    = (const float*)d_in[5];
    const float* b_col    = (const float*)d_in[6];
    const float* HG_up    = (const float*)d_in[7];   // U x P
    const float* HG_pu    = (const float*)d_in[8];   // P x U
    const float* HG_src   = (const float*)d_in[9];   // P x E
    const float* HG_tar   = (const float*)d_in[10];  // E x P
    const float* geo_grph = (const float*)d_in[11];  // P x P
    const int*   user_idx = (const int*)d_in[12];
    float* out = (float*)d_out;

    // Output layout (floats): hg_pois | geo_pois | trans_pois | hg_users | geo_users
    float* hg_pois    = out;
    float* geo_pois   = out + (size_t)P_DIM * D_DIM;
    float* trans_pois = out + (size_t)2 * P_DIM * D_DIM;
    float* hg_users   = out + (size_t)3 * P_DIM * D_DIM;
    float* geo_users  = hg_users + (size_t)B_DIM * D_DIM;

    // Workspace carve (256B aligned slices)
    char* wsp = (char*)d_ws;
    auto alloc = [&](size_t bytes) -> void* {
        void* p = (void*)wsp;
        wsp += (bytes + 255) & ~(size_t)255;
        return p;
    };
    __bf16* colT  = (__bf16*)alloc((size_t)2 * D_DIM * P_DIM);  // gates^T (128 x P)
    __bf16* geoT  = (__bf16*)alloc((size_t)2 * D_DIM * P_DIM);
    __bf16* seqT  = (__bf16*)alloc((size_t)2 * D_DIM * P_DIM);
    __bf16* hgT   = (__bf16*)alloc((size_t)2 * D_DIM * P_DIM);  // hg_pois^T
    __bf16* gpT   = (__bf16*)alloc((size_t)2 * D_DIM * P_DIM);  // geo_pois^T
    __bf16* t1T   = (__bf16*)alloc((size_t)2 * D_DIM * U_DIM);  // (HG_up@col)^T
    __bf16* t2T   = (__bf16*)alloc((size_t)2 * D_DIM * U_DIM);  // (HG_tar@seq)^T
    float*  col_f = (float*)alloc((size_t)4 * P_DIM * D_DIM);
    float*  geo_f = (float*)alloc((size_t)4 * P_DIM * D_DIM);
    float*  seq_f = (float*)alloc((size_t)4 * P_DIM * D_DIM);
    float*  u1    = (float*)alloc((size_t)4 * U_DIM * D_DIM);
    float*  u2    = (float*)alloc((size_t)4 * U_DIM * D_DIM);
    __bf16* wtG   = (__bf16*)alloc((size_t)2 * D_DIM * D_DIM);
    __bf16* wtS   = (__bf16*)alloc((size_t)2 * D_DIM * D_DIM);
    __bf16* wtC   = (__bf16*)alloc((size_t)2 * D_DIM * D_DIM);

    const float ALPHA_HALF = 0.4f;   // ALPHA/2 with T_GEO=1

    // 1) W -> W^T bf16
    wt_convert_kernel<<<64, 256, 0, stream>>>(w_geo, wtG);
    wt_convert_kernel<<<64, 256, 0, stream>>>(w_seq, wtS);
    wt_convert_kernel<<<64, 256, 0, stream>>>(w_col, wtC);

    // 2) gates (WMMA, K=128), fp32 + bf16^T outputs
    gate_kernel<<<P_DIM / 128, 256, 0, stream>>>(poi, wtG, b_geo, geo_f, geoT, P_DIM);
    gate_kernel<<<P_DIM / 128, 256, 0, stream>>>(poi, wtS, b_seq, seq_f, seqT, P_DIM);
    gate_kernel<<<P_DIM / 128, 256, 0, stream>>>(poi, wtC, b_col, col_f, colT, P_DIM);

    // 3) t1 = HG_up @ col_gate           (4096 x 8192) -> bf16^T only
    gemm_bf16_kernel<<<U_DIM / 128, 256, 0, stream>>>(
        HG_up, colT, P_DIM, 1.0f, nullptr, 0.0f, nullptr, t1T, U_DIM);
    // 4) hg_pois = col + HG_pu @ t1      (8192 x 4096)
    gemm_bf16_kernel<<<P_DIM / 128, 256, 0, stream>>>(
        HG_pu, t1T, U_DIM, 1.0f, col_f, 1.0f, hg_pois, hgT, P_DIM);
    // 5) geo_pois = geo + 0.4 * geo_graph @ geo   (8192 x 8192)
    gemm_bf16_kernel<<<P_DIM / 128, 256, 0, stream>>>(
        geo_grph, geoT, P_DIM, ALPHA_HALF, geo_f, 1.0f, geo_pois, gpT, P_DIM);
    // 6) t2 = HG_tar @ seq_gate          (4096 x 8192) -> bf16^T only
    gemm_bf16_kernel<<<U_DIM / 128, 256, 0, stream>>>(
        HG_tar, seqT, P_DIM, 1.0f, nullptr, 0.0f, nullptr, t2T, U_DIM);
    // 7) trans_pois = seq + HG_src @ t2  (8192 x 4096)
    gemm_bf16_kernel<<<P_DIM / 128, 256, 0, stream>>>(
        HG_src, t2T, U_DIM, 1.0f, seq_f, 1.0f, trans_pois, nullptr, P_DIM);
    // 8) u1 = HG_up @ hg_pois ; u2 = HG_up @ geo_pois   (4096 x 8192)
    gemm_bf16_kernel<<<U_DIM / 128, 256, 0, stream>>>(
        HG_up, hgT, P_DIM, 1.0f, nullptr, 0.0f, u1, nullptr, U_DIM);
    gemm_bf16_kernel<<<U_DIM / 128, 256, 0, stream>>>(
        HG_up, gpT, P_DIM, 1.0f, nullptr, 0.0f, u2, nullptr, U_DIM);

    // 9) gather user rows
    gather_kernel<<<(B_DIM * D_DIM) / 256, 256, 0, stream>>>(
        u1, u2, user_idx, hg_users, geo_users);
}
